// PatchGAT_10625749090914
// MI455X (gfx1250) — matrix operand order repaired and checked by reading, not compiled, not run on hardware
//
#include <hip/hip_runtime.h>

// ---------------------------------------------------------------------------
// Types for CDNA5 WMMA (wave32, 16x16x32 bf16 -> f32)
// ---------------------------------------------------------------------------
typedef __attribute__((ext_vector_type(16))) __bf16 v16bf;
typedef __attribute__((ext_vector_type(8)))  __bf16 v8bf;
typedef __attribute__((ext_vector_type(8)))  float  v8f;

union V16U { v16bf v; v8bf h[2]; };

#define GAT_N 50000
#define GAT_E 800000

__device__ __forceinline__ float lrelu(float x, float s) {
    return x >= 0.0f ? x : s * x;
}

// Ordered-uint encoding of float for atomicMax on unsigned
__device__ __forceinline__ unsigned ford_enc(float f) {
    unsigned u = __float_as_uint(f);
    return (u & 0x80000000u) ? ~u : (u | 0x80000000u);
}
__device__ __forceinline__ float ford_dec(unsigned u) {
    unsigned v = (u & 0x80000000u) ? (u & 0x7fffffffu) : ~u;
    return __uint_as_float(v);
}

// ---------------------------------------------------------------------------
// f32 -> bf16 (bits) with round-to-nearest-even
// ---------------------------------------------------------------------------
__global__ void k_f32_to_bf16(const float* __restrict__ in,
                              unsigned short* __restrict__ out, int n) {
    int t = blockIdx.x * blockDim.x + threadIdx.x;
    if (t >= n) return;
    unsigned u = __float_as_uint(in[t]);
    unsigned r = u + 0x7fffu + ((u >> 16) & 1u);
    out[t] = (unsigned short)(r >> 16);
}

// ---------------------------------------------------------------------------
// WMMA GEMM: C[M,Nc] = A[M,K] * B, bf16 inputs, f32 out.
// A row-major. B element (k,n) at B[k*bk + n*bn] (supports W and W^T).
// Block = 256 threads = 8 waves; C block tile 32x64; 16x16 tile per wave;
// K step 64 -> two v_wmma_f32_16x16x32_bf16 per wave per barrier pair.
// Requires: K % 64 == 0, Nc % 64 == 0, M*Nc < 2^31.
// ---------------------------------------------------------------------------
__global__ __launch_bounds__(256) void k_gemm_bf16_wmma(
    const unsigned short* __restrict__ A,
    const unsigned short* __restrict__ B,
    float* __restrict__ C,
    int M, int K, int Nc, int bk, int bn)
{
    __shared__ alignas(16) unsigned short As[32][72];  // [m][k], row 144B (16B mult)
    __shared__ alignas(16) unsigned short Bs[64][72];  // [n][k], transposed in LDS

    const int tid  = threadIdx.x;
    const int wave = tid >> 5;
    const int lane = tid & 31;
    const int half = lane >> 4;
    const int l16  = lane & 15;
    const int wm   = wave & 1;   // row tile 0..1 (16 rows each)
    const int wn   = wave >> 1;  // col tile 0..3 (16 cols each)
    const int mbase = blockIdx.x * 32;
    const int nbase = blockIdx.y * 64;

    v8f acc = {};

    for (int kb = 0; kb < K; kb += 64) {
        // ---- stage A tile 32x64: one 16B vector per thread ----
        {
            int l = tid * 8;
            int r = l >> 6;
            int c = l & 63;
            int gr = mbase + r;
            if (gr >= M) gr = M - 1;  // clamp; stores are predicated
            const uint4* sp = (const uint4*)(A + (size_t)gr * K + (kb + c));
            *(uint4*)&As[r][c] = *sp;
        }
        // ---- stage B tile 64(k) x 64(n), store transposed Bs[n][k] ----
#pragma unroll
        for (int i = 0; i < 16; ++i) {
            int l = i * 256 + tid;
            int k = l >> 6;
            int n = l & 63;
            Bs[n][k] = B[(size_t)(kb + k) * bk + (size_t)(nbase + n) * bn];
        }
        __syncthreads();

        // ---- build fragments per ISA 7.12.2 layouts ----
        V16U a0, a1, b0, b1;
        {
            int am  = wm * 16 + l16;      // A: lane holds row M = lane%16
            int ak0 = 8 * half;           // elems 0..7 -> K = 8*half + e
            a0.h[0] = *(const v8bf*)&As[am][ak0];
            a0.h[1] = *(const v8bf*)&As[am][ak0 + 16];       // elems 8..15 -> +16
            a1.h[0] = *(const v8bf*)&As[am][32 + ak0];
            a1.h[1] = *(const v8bf*)&As[am][32 + ak0 + 16];
        }
        {
            int br  = wn * 16 + l16;      // B: lane holds col N = lane%16
            int bk0 = 8 * half;           // elems 0..7 -> K = 8*half + e, 8..15 -> +16
            b0.h[0] = *(const v8bf*)&Bs[br][bk0];
            b0.h[1] = *(const v8bf*)&Bs[br][bk0 + 16];
            b1.h[0] = *(const v8bf*)&Bs[br][32 + bk0];
            b1.h[1] = *(const v8bf*)&Bs[br][32 + bk0 + 16];
        }

        acc = __builtin_amdgcn_wmma_f32_16x16x32_bf16(
            false, a0.v, false, b0.v, (short)0, acc, false, false);
        acc = __builtin_amdgcn_wmma_f32_16x16x32_bf16(
            false, a1.v, false, b1.v, (short)0, acc, false, false);

        __syncthreads();
    }

    // ---- store D: element r -> M = 8*half + r, N = lane%16 (32-bit indexing) ----
    const int n    = nbase + wn * 16 + l16;
    const int mrow = mbase + wm * 16 + half * 8;
    int idx = mrow * Nc + n;
    if (mbase + 32 <= M) {
        // fast path: whole tile in range, unpredicated stores
#pragma unroll
        for (int r = 0; r < 8; ++r) C[idx + r * Nc] = acc[r];
    } else {
#pragma unroll
        for (int r = 0; r < 8; ++r) {
            if (mrow + r < M) C[idx + r * Nc] = acc[r];
        }
    }
}

// ---------------------------------------------------------------------------
// el/er: per (node, head) dot of z[n,h,:] with attn_l / attn_r
// ---------------------------------------------------------------------------
__global__ void k_elr(const float* __restrict__ z,
                      const float* __restrict__ al,
                      const float* __restrict__ ar,
                      float* __restrict__ el, float* __restrict__ er,
                      int Nn, int D, int Hd) {
    int t = blockIdx.x * blockDim.x + threadIdx.x;
    if (t >= Nn * 4) return;
    int n = t >> 2, h = t & 3;
    const float* zp = z + (size_t)n * Hd + h * D;
    const float* lp = al + h * D;
    const float* rp = ar + h * D;
    float sl = 0.f, sr = 0.f;
    for (int d = 0; d < D; ++d) {
        float v = zp[d];
        sl += v * lp[d];
        sr += v * rp[d];
    }
    el[t] = sl;
    er[t] = sr;
}

__global__ void k_fill_u32(unsigned* __restrict__ p, unsigned v, int n) {
    int t = blockIdx.x * blockDim.x + threadIdx.x;
    if (t < n) p[t] = v;
}
__global__ void k_fill_f32(float* __restrict__ p, float v, int n) {
    int t = blockIdx.x * blockDim.x + threadIdx.x;
    if (t < n) p[t] = v;
}
// agg[n, j] = bias[j]
__global__ void k_init_bias(float* __restrict__ agg, const float* __restrict__ b,
                            int total, int hdmask) {
    int t = blockIdx.x * blockDim.x + threadIdx.x;
    if (t < total) agg[t] = b[t & hdmask];
}

// pass 1: segment max of leakyrelu(el[src]+er[dst], 0.2) over dst
__global__ void k_edge_max(const int* __restrict__ src, const int* __restrict__ dst,
                           const float* __restrict__ el, const float* __restrict__ er,
                           unsigned* __restrict__ emax, int Ee) {
    int t = blockIdx.x * blockDim.x + threadIdx.x;
    if (t >= Ee * 4) return;
    int e = t >> 2, h = t & 3;
    int s = src[e], d = dst[e];
    float v = lrelu(el[s * 4 + h] + er[d * 4 + h], 0.2f);
    atomicMax(&emax[d * 4 + h], ford_enc(v));
}

// pass 2: ex = exp(e - max); segment sum over dst
__global__ void k_edge_exp(const int* __restrict__ src, const int* __restrict__ dst,
                           const float* __restrict__ el, const float* __restrict__ er,
                           const unsigned* __restrict__ emax,
                           float* __restrict__ ex, float* __restrict__ esum, int Ee) {
    int t = blockIdx.x * blockDim.x + threadIdx.x;
    if (t >= Ee * 4) return;
    int e = t >> 2, h = t & 3;
    int s = src[e], d = dst[e];
    float v = lrelu(el[s * 4 + h] + er[d * 4 + h], 0.2f);
    float m = ford_dec(emax[d * 4 + h]);
    float xv = __expf(v - m);
    ex[t] = xv;
    atomicAdd(&esum[d * 4 + h], xv);
}

// pass 3: agg[dst, :] += (ex/esum) * z[src, :]
__global__ void k_edge_scatter(const int* __restrict__ src, const int* __restrict__ dst,
                               const float* __restrict__ z,
                               const float* __restrict__ ex,
                               const float* __restrict__ esum,
                               float* __restrict__ agg,
                               int total, int hdshift, int hdmask, int dshift) {
    int t = blockIdx.x * blockDim.x + threadIdx.x;
    if (t >= total) return;
    int e = t >> hdshift;
    int j = t & hdmask;
    int s = src[e], d = dst[e];
    int h = j >> dshift;
    float alpha = ex[e * 4 + h] / esum[d * 4 + h];
    int Hd = hdmask + 1;
    atomicAdd(&agg[(size_t)d * Hd + j], alpha * z[(size_t)s * Hd + j]);
}

__global__ void k_lrelu(float* __restrict__ x, float s, int n) {
    int t = blockIdx.x * blockDim.x + threadIdx.x;
    if (t < n) x[t] = lrelu(x[t], s);
}
// x[n, c] = lrelu(x[n,c] + b[c], s); cols power of two
__global__ void k_bias_lrelu(float* __restrict__ x, const float* __restrict__ b,
                             int total, int cmask, float s) {
    int t = blockIdx.x * blockDim.x + threadIdx.x;
    if (t < total) x[t] = lrelu(x[t] + b[t & cmask], s);
}

// out[n,o] = lrelu(mean_h lrelu(agg[n, h*64+o], .01), .01)
__global__ void k_head_mean(const float* __restrict__ agg, float* __restrict__ out,
                            int Nn) {
    int t = blockIdx.x * blockDim.x + threadIdx.x;
    if (t >= Nn * 64) return;
    int n = t >> 6, o = t & 63;
    const float* p = agg + (size_t)n * 256 + o;
    float s = 0.f;
#pragma unroll
    for (int h = 0; h < 4; ++h) s += lrelu(p[h * 64], 0.01f);
    out[t] = lrelu(s * 0.25f, 0.01f);
}

// ---------------------------------------------------------------------------
static inline unsigned cdiv(long long a, long long b) { return (unsigned)((a + b - 1) / b); }

extern "C" void kernel_launch(void* const* d_in, const int* in_sizes, int n_in,
                              void* d_out, int out_size, void* d_ws, size_t ws_size,
                              hipStream_t stream) {
    (void)in_sizes; (void)n_in; (void)out_size; (void)ws_size;
    const int Nn = GAT_N, Ee = GAT_E;

    const float* n_feat = (const float*)d_in[0];
    const int*   src    = (const int*)d_in[1];
    const int*   dst    = (const int*)d_in[2];
    const float* g0_W  = (const float*)d_in[3];
    const float* g0_al = (const float*)d_in[4];
    const float* g0_ar = (const float*)d_in[5];
    const float* g0_b  = (const float*)d_in[6];
    const float* d0_W  = (const float*)d_in[7];
    const float* d0_b  = (const float*)d_in[8];
    const float* g1_W  = (const float*)d_in[9];
    const float* g1_al = (const float*)d_in[10];
    const float* g1_ar = (const float*)d_in[11];
    const float* g1_b  = (const float*)d_in[12];
    const float* d1_W  = (const float*)d_in[13];
    const float* d1_b  = (const float*)d_in[14];
    const float* g2_W  = (const float*)d_in[15];
    const float* g2_al = (const float*)d_in[16];
    const float* g2_ar = (const float*)d_in[17];
    const float* g2_b  = (const float*)d_in[18];

    // workspace carve-out
    char* p = (char*)d_ws;
    auto take = [&](size_t bytes) -> void* {
        void* r = (void*)p;
        p += (bytes + 255) & ~(size_t)255;
        return r;
    };
    float*          zf   = (float*)take((size_t)Nn * 512 * 4);   // GEMM out / z
    float*          agg  = (float*)take((size_t)Nn * 512 * 4);   // aggregation
    unsigned short* abf  = (unsigned short*)take((size_t)Nn * 512 * 2);
    unsigned short* wbf  = (unsigned short*)take((size_t)65536 * 2);
    float*          el   = (float*)take((size_t)Nn * 4 * 4);
    float*          er   = (float*)take((size_t)Nn * 4 * 4);
    unsigned*       emax = (unsigned*)take((size_t)Nn * 4 * 4);
    float*          esum = (float*)take((size_t)Nn * 4 * 4);
    float*          ex   = (float*)take((size_t)Ee * 4 * 4);

    const unsigned NEG_INF_ORD = 0x007FFFFFu;  // ford_enc(-inf)

    auto run_gat = [&](const float* X, int Fin, int D,
                       const float* gW, const float* al, const float* ar,
                       const float* gb) {
        int Hd = 4 * D;
        int nx = Nn * Fin;
        int nw = Fin * Hd;
        k_f32_to_bf16<<<cdiv(nx, 256), 256, 0, stream>>>(X, abf, nx);
        k_f32_to_bf16<<<cdiv(nw, 256), 256, 0, stream>>>(gW, wbf, nw);
        dim3 g(cdiv(Nn, 32), (unsigned)(Hd / 64));
        k_gemm_bf16_wmma<<<g, 256, 0, stream>>>(abf, wbf, zf, Nn, Fin, Hd,
                                                /*bk=*/Hd, /*bn=*/1);
        k_elr<<<cdiv(Nn * 4, 256), 256, 0, stream>>>(zf, al, ar, el, er, Nn, D, Hd);
        k_fill_u32<<<cdiv(Nn * 4, 256), 256, 0, stream>>>(emax, NEG_INF_ORD, Nn * 4);
        k_fill_f32<<<cdiv(Nn * 4, 256), 256, 0, stream>>>(esum, 0.0f, Nn * 4);
        k_edge_max<<<cdiv((long long)Ee * 4, 256), 256, 0, stream>>>(src, dst, el, er, emax, Ee);
        k_edge_exp<<<cdiv((long long)Ee * 4, 256), 256, 0, stream>>>(src, dst, el, er, emax, ex, esum, Ee);
        int totN = Nn * Hd;
        k_init_bias<<<cdiv(totN, 256), 256, 0, stream>>>(agg, gb, totN, Hd - 1);
        int hdshift = (Hd == 512) ? 9 : 8;
        int dshift  = (D == 128) ? 7 : 6;
        int totE = Ee * Hd;  // <= 409.6M, fits int
        k_edge_scatter<<<cdiv(totE, 256), 256, 0, stream>>>(
            src, dst, zf, ex, esum, agg, totE, hdshift, Hd - 1, dshift);
    };

    auto run_dense = [&](const float* dW, const float* db) {
        // in: agg [N,512] (post-activation), out: zf [N,128] = agg @ dW^T + db
        int nx = Nn * 512;
        k_f32_to_bf16<<<cdiv(nx, 256), 256, 0, stream>>>(agg, abf, nx);
        k_f32_to_bf16<<<cdiv(65536, 256), 256, 0, stream>>>(dW, wbf, 65536);
        dim3 g(cdiv(Nn, 32), 2);
        // B[k,n] = dW[n,k] -> bk=1, bn=512
        k_gemm_bf16_wmma<<<g, 256, 0, stream>>>(abf, wbf, zf, Nn, 512, 128,
                                                /*bk=*/1, /*bn=*/512);
        k_bias_lrelu<<<cdiv(Nn * 128, 256), 256, 0, stream>>>(zf, db, Nn * 128, 127, 0.01f);
    };

    // ---- layer 0 ----
    run_gat(n_feat, 128, 128, g0_W, g0_al, g0_ar, g0_b);
    k_lrelu<<<cdiv(Nn * 512, 256), 256, 0, stream>>>(agg, 0.01f, Nn * 512);
    run_dense(d0_W, d0_b);                 // -> zf [N,128]
    // ---- layer 1 ----
    run_gat(zf, 128, 128, g1_W, g1_al, g1_ar, g1_b);
    k_lrelu<<<cdiv(Nn * 512, 256), 256, 0, stream>>>(agg, 0.01f, Nn * 512);
    run_dense(d1_W, d1_b);                 // -> zf [N,128]
    // ---- layer 2 (OUT=64) ----
    run_gat(zf, 128, 64, g2_W, g2_al, g2_ar, g2_b);
    k_head_mean<<<cdiv(Nn * 64, 256), 256, 0, stream>>>(agg, (float*)d_out, Nn);
}
